// VectorQuantizer_53343493816934
// MI455X (gfx1250) — matrix operand order repaired and checked by reading, not compiled, not run on hardware
//
#include <hip/hip_runtime.h>

// Problem constants (from reference)
#define NROWS   65536          // 64*32*32 flattened z rows
#define DIM     64             // embedding dim
#define NCODES  1024           // codebook size
#define SCALE   1024.0f        // pre-scale for f16 codebook (values -> [-1,1])
#define ROWSTR  72             // padded LDS row stride (halves): conflict-free b128
#define TILES   4              // 16-row tiles per wave
#define WAVES   8              // waves per block (256 threads, wave32)
#define ROWS_PER_BLOCK (WAVES*TILES*16)   // 512
#define NBLOCKS (NROWS/ROWS_PER_BLOCK)    // 128

typedef __attribute__((ext_vector_type(16))) _Float16 v16h;
typedef __attribute__((ext_vector_type(8)))  float    v8f;

union B16 { uint4 u[2]; v16h h; };

// ---- prep: f32 codebook -> scaled f16 codebook -------------------------------
__global__ void vq_prep_convert(const float* __restrict__ emb,
                                _Float16* __restrict__ embH) {
  int id = blockIdx.x * 256 + threadIdx.x;            // 65536 elements
  embH[id] = (_Float16)(emb[id] * SCALE);
}

// ---- prep: scaled squared norms  e2s[k] = SCALE * ||e_k||^2 ------------------
__global__ void vq_prep_norms(const float* __restrict__ emb,
                              float* __restrict__ e2s) {
  int r = blockIdx.x * 256 + threadIdx.x;             // 1024 rows
  if (r < NCODES) {
    const float* e = emb + (size_t)r * DIM;
    float s = 0.f;
#pragma unroll 8
    for (int k = 0; k < DIM; ++k) { float v = e[k]; s += v * v; }
    e2s[r] = s * SCALE;
  }
}

// ---- main: WMMA distances + argmin + gather + loss partials ------------------
__global__ __launch_bounds__(256) void vq_main(
    const float* __restrict__ z, const float* __restrict__ embF,
    const _Float16* __restrict__ embH, const float* __restrict__ e2s,
    float* __restrict__ outQ, float* __restrict__ partial) {
  __shared__ _Float16 sEmb[NCODES * ROWSTR];   // 144 KB padded f16 codebook
  __shared__ float    sE2[NCODES];             // 4 KB scaled norms
  __shared__ float    sWsum[WAVES];

  const int tid  = threadIdx.x;
  const int lane = tid & 31;
  const int w    = tid >> 5;
  const int n15  = lane & 15;                  // N (col) / M (row) sub-index
  const int kh   = lane >> 4;                  // K half selector

  // Stage codebook into LDS with padded rows (16B chunks, coalesced)
  for (int it = tid; it < NCODES * DIM / 8; it += 256) {
    int r = it >> 3, c8 = it & 7;
    uint4 v = ((const uint4*)embH)[it];
    *((uint4*)(sEmb + r * ROWSTR + c8 * 8)) = v;
  }
  for (int it = tid; it < NCODES; it += 256) sE2[it] = e2s[it];
  __syncthreads();

  const int waveRow0 = blockIdx.x * ROWS_PER_BLOCK + w * (TILES * 16);

  // Build A fragments: 16-bit A 16x32 layout.
  // lane<16: K = 8*kh..+7 (v0..3) and 16+8*kh..+7 (v4..7), per t-step (+32t)
  v16h A[TILES][2];
#pragma unroll
  for (int i = 0; i < TILES; ++i) {
    const float* zr = z + (size_t)(waveRow0 + i * 16 + n15) * DIM;
#pragma unroll
    for (int t = 0; t < 2; ++t) {
      const int k0 = t * 32 + kh * 8;
      float4 f0 = *(const float4*)(zr + k0);
      float4 f1 = *(const float4*)(zr + k0 + 4);
      float4 f2 = *(const float4*)(zr + k0 + 16);
      float4 f3 = *(const float4*)(zr + k0 + 20);
      v16h a;
      a[0]  = (_Float16)f0.x; a[1]  = (_Float16)f0.y;
      a[2]  = (_Float16)f0.z; a[3]  = (_Float16)f0.w;
      a[4]  = (_Float16)f1.x; a[5]  = (_Float16)f1.y;
      a[6]  = (_Float16)f1.z; a[7]  = (_Float16)f1.w;
      a[8]  = (_Float16)f2.x; a[9]  = (_Float16)f2.y;
      a[10] = (_Float16)f2.z; a[11] = (_Float16)f2.w;
      a[12] = (_Float16)f3.x; a[13] = (_Float16)f3.y;
      a[14] = (_Float16)f3.z; a[15] = (_Float16)f3.w;
      A[i][t] = a;
    }
  }

  float minv[TILES][8];
  int   mini[TILES][8];
#pragma unroll
  for (int i = 0; i < TILES; ++i)
#pragma unroll
    for (int j = 0; j < 8; ++j) { minv[i][j] = 3.4e38f; mini[i][j] = 0; }

  // Sweep all 1024 codes in groups of 16 (B from LDS, K=64 via 2 WMMAs)
  for (int g = 0; g < NCODES / 16; ++g) {
    const _Float16* bs = sEmb + (g * 16 + n15) * ROWSTR + kh * 8;
    B16 b0, b1;
    b0.u[0] = *((const uint4*)(bs));
    b0.u[1] = *((const uint4*)(bs + 16));
    b1.u[0] = *((const uint4*)(bs + 32));
    b1.u[1] = *((const uint4*)(bs + 48));
    const float e2v = sE2[g * 16 + n15];

#pragma unroll
    for (int i = 0; i < TILES; ++i) {
      v8f c = {};
      c = __builtin_amdgcn_wmma_f32_16x16x32_f16(false, A[i][0], false, b0.h,
                                                 (short)0, c, false, false);
      c = __builtin_amdgcn_wmma_f32_16x16x32_f16(false, A[i][1], false, b1.h,
                                                 (short)0, c, false, false);
#pragma unroll
      for (int j = 0; j < 8; ++j) {
        float d = e2v - 2.0f * c[j];   // SCALE*||e||^2 - 2*z.(SCALE*e)
        if (d < minv[i][j]) { minv[i][j] = d; mini[i][j] = g * 16 + n15; }
      }
    }
  }

  // Argmin across the 16 lanes of each half (rows 0-7 in lanes 0-15, 8-15 in 16-31)
  float lossacc = 0.f;
#pragma unroll
  for (int i = 0; i < TILES; ++i) {
#pragma unroll
    for (int j = 0; j < 8; ++j) {
      float v = minv[i][j];
      int   ix = mini[i][j];
#pragma unroll
      for (int m = 1; m < 16; m <<= 1) {
        float ov = __shfl_xor(v, m, 16);
        int   oi = __shfl_xor(ix, m, 16);
        if (ov < v || (ov == v && oi < ix)) { v = ov; ix = oi; }
      }
      minv[i][j] = v; mini[i][j] = ix;
    }
    // Gather winning codebook rows (f32) and write quantized output
    const int rowBase = waveRow0 + i * 16;
#pragma unroll
    for (int m = 0; m < 16; ++m) {
      int idx = __shfl(mini[i][m & 7], (m >> 3) * 16, 32);
      float2 q = ((const float2*)(embF + (size_t)idx * DIM))[lane];
      size_t off = (size_t)(rowBase + m) * DIM;
      float2 zv = ((const float2*)(z + off))[lane];
      ((float2*)(outQ + off))[lane] = q;
      float dx = q.x - zv.x, dy = q.y - zv.y;
      lossacc += dx * dx + dy * dy;
    }
  }

  // Deterministic loss partial per block: wave reduce -> LDS -> sequential sum
#pragma unroll
  for (int m = 16; m >= 1; m >>= 1) lossacc += __shfl_xor(lossacc, m, 32);
  if (lane == 0) sWsum[w] = lossacc;
  __syncthreads();
  if (tid == 0) {
    float s = 0.f;
    for (int k = 0; k < WAVES; ++k) s += sWsum[k];
    partial[blockIdx.x] = s;
  }
}

// ---- finish: reduce 128 partials -> loss scalar ------------------------------
__global__ void vq_finish(const float* __restrict__ partial,
                          float* __restrict__ out) {
  if (threadIdx.x == 0 && blockIdx.x == 0) {
    float s = 0.f;
    for (int i = 0; i < NBLOCKS; ++i) s += partial[i];
    // loss = (BETA + 1) * mean((q - z)^2) = 1.25 * sum / (N*D)
    out[(size_t)NROWS * DIM] = 1.25f * (s / (float)((size_t)NROWS * DIM));
  }
}

extern "C" void kernel_launch(void* const* d_in, const int* in_sizes, int n_in,
                              void* d_out, int out_size, void* d_ws, size_t ws_size,
                              hipStream_t stream) {
  (void)in_sizes; (void)n_in; (void)out_size; (void)ws_size;
  const float* z   = (const float*)d_in[0];   // [64,32,32,64] f32
  const float* emb = (const float*)d_in[1];   // [1024,64] f32
  float* out = (float*)d_out;                 // [65536*64] quantized + [1] loss

  _Float16* embH = (_Float16*)d_ws;                                   // 128 KB
  float* e2s     = (float*)((char*)d_ws + (size_t)NCODES * DIM * 2);  // 4 KB
  float* partial = e2s + NCODES;                                      // 512 B

  vq_prep_convert<<<NCODES * DIM / 256, 256, 0, stream>>>(emb, embH);
  vq_prep_norms<<<(NCODES + 255) / 256, 256, 0, stream>>>(emb, e2s);
  vq_main<<<NBLOCKS, 256, 0, stream>>>(z, emb, embH, e2s, out, partial);
  vq_finish<<<1, 32, 0, stream>>>(partial, out);
}
